// ConvLayer_79817672229344
// MI455X (gfx1250) — matrix operand order repaired and checked by reading, not compiled, not run on hardware
//
#include <hip/hip_runtime.h>
#include <cstdint>
#include <cstddef>

// ---------------------------------------------------------------------------
// Fused stride-2 transposed conv (3x3, 512->256) + 4x4 FIR + bias/leaky/clamp
// MI455X (gfx1250): wave32 WMMA f16 16x16x32 (f32 acc), async global->LDS
// staging (ASYNCcnt) double-buffered in 320KB WGP LDS, FIR overlapped with
// the next chunk's async stage.
// ---------------------------------------------------------------------------

typedef _Float16 v8h  __attribute__((ext_vector_type(8)));
typedef _Float16 v16h __attribute__((ext_vector_type(16)));
typedef float    v8f  __attribute__((ext_vector_type(8)));

#define IC 512
#define OC 256
#define IH 64
#define IW 64
#define OH 128
#define OW 128

#define OT 28                  // output tile (28x28) -> y tile 32x32, classes 16x16
#define XT 17                  // x tile side (17x17 input pixels)
#define CC 128                 // channels staged per c-pass
#define XPITCH 136             // f16 per pixel in LDS (128c + pad; 272B, 16B aligned)
#define ROWSTEP (XT * XPITCH * 2)   // 4624 bytes per iy row in x stage

#define XS_BYTES (XT * XT * XPITCH * 2)   // 78608
#define XS0_OFF 0
#define XS1_OFF XS_BYTES                   // 78608
#define H_OFF   XS1_OFF                    // FIR h-pass (f16, 55552B) aliases xs1 only
#define Y_OFF   (2 * XS_BYTES)             // 157216

#define YP 33                  // y LDS pitch (dwords) per ox (odd => conflict-free)
#define YROW (32 * YP)         // 1056
#define YLDS_DW (32 * YROW)    // 33792
#define YLDS_BYTES (YLDS_DW * 4)              // 135168
#define SMEM_MAIN (Y_OFF + YLDS_BYTES)        // 292384  (< 320KB)

#define XP_PITCH 520           // prep_x LDS pitch (halves per ix)
#define SMEM_PREPX (64 * XP_PITCH * 2)        // 66560

#define WSCALE  0.0147313913f  // 1/sqrt(9*512)
#define ACTGAIN 1.41421356237f

static __device__ __forceinline__ v16h cat16(v8h lo, v8h hi) {
    return __builtin_shufflevector(lo, hi, 0,1,2,3,4,5,6,7,8,9,10,11,12,13,14,15);
}

// ---------------------------------------------------------------------------
// prep_w: f32 OIHW -> f16 B-fragments in exact per-lane order.
// Fragment (tap, cblk32, ocblk16) = 32 lanes x 32B (1KB contiguous).
// ---------------------------------------------------------------------------
__global__ __launch_bounds__(256) void prep_w_kernel(const float* __restrict__ w,
                                                     _Float16* __restrict__ wp) {
    int t = blockIdx.x * 256 + threadIdx.x;        // < 589824
    int i   = t & 7;
    int ln  = (t >> 3) & 31;
    int ocb = (t >> 8) & 15;
    int cb  = (t >> 12) & 15;
    int tap = t >> 16;                              // 0..8
    int ky = tap / 3, kx = tap % 3;
    int oc = ocb * 16 + (ln & 15);
    int c  = cb * 32 + (i >> 2) * 16 + (ln >> 4) * 8 + 2 * (i & 3);
    size_t base = (((size_t)oc * IC + c) * 3 + ky) * 3 + kx;
    wp[(size_t)t * 2]     = (_Float16)(w[base] * WSCALE);
    wp[(size_t)t * 2 + 1] = (_Float16)(w[base + 9] * WSCALE);
}

// ---------------------------------------------------------------------------
// prep_x: NCHW f32 -> NHWC f16 (x16[n][iy][ix][c]), LDS-bounced transpose.
// ---------------------------------------------------------------------------
__global__ __launch_bounds__(256) void prep_x_kernel(const float* __restrict__ x,
                                                     _Float16* __restrict__ x16) {
    extern __shared__ char pmem[];
    _Float16* lx = (_Float16*)pmem;
    const int iy = blockIdx.x;
    const int n  = blockIdx.y;
    const int tid = threadIdx.x;
    const int ix = tid & 63;
    const int c0 = tid >> 6;
    for (int k = 0; k < 128; ++k) {
        int c = (k << 2) | c0;
        float v = x[(((size_t)n * IC + c) * IH + iy) * IW + ix];
        lx[ix * XP_PITCH + c] = (_Float16)v;
    }
    __syncthreads();
    const size_t gbytes = ((size_t)(n * 64 + iy) * 64) * 512 * 2;
    for (int k = 0; k < 16; ++k) {
        int t = (k << 8) | tid;
        int ixw = t >> 6;
        int co  = t & 63;
        *(int4*)((char*)x16 + gbytes + (size_t)ixw * 1024 + co * 16) =
            *(const int4*)((const char*)lx + ixw * (XP_PITCH * 2) + co * 16);
    }
}

// ---------------------------------------------------------------------------
// Main fused kernel. Grid: (25 tiles, 16 batch, 2 oc-halves). 256 thr = 8 waves.
// ---------------------------------------------------------------------------
__global__ __launch_bounds__(256) void conv_up_fir_kernel(
    const _Float16* __restrict__ x16, const _Float16* __restrict__ wp,
    const float* __restrict__ bias, float* __restrict__ out)
{
    extern __shared__ char smem[];
    float*    yL = (float*)(smem + Y_OFF);
    _Float16* hH = (_Float16*)(smem + H_OFF);

    const int tid = threadIdx.x;
    const int ln  = tid & 31;
    const int wv  = tid >> 5;
    const int wm  = wv & 3;
    const int oct = wv >> 2;

    const int tile = blockIdx.x;
    const int ty = tile / 5, tx = tile % 5;
    const int n  = blockIdx.y;
    const int zh = blockIdx.z;

    const int r0 = OT * ty, c0 = OT * tx;
    const int iy0 = 14 * ty - 2, ix0 = 14 * tx - 2;

    // wave tap assignment (36 wmma / kstep / wave, perfectly balanced)
    const int py  = (wm == 3) ? 1 : 0;
    const int px  = (wm == 2) ? 1 : 0;
    const int ky0 = (wm == 1) ? 2 : ((wm == 3) ? 1 : 0);
    const int ky1 = (wm == 2) ? 2 : ky0;
    const int kx0 = (wm == 2) ? 1 : 0;
    const int kx1 = (wm == 2) ? 1 : 2;
    const int rB0 = wm * 4;

    const int lane16 = ln & 15;
    const int hiL    = ln >> 4;
    const int laneAoff = lane16 * (XPITCH * 2) + hiL * 16;
    const int ocl = oct * 16 + lane16;

    const unsigned long long xnbase =
        (unsigned long long)(uintptr_t)x16 + ((unsigned long long)n << 22);
    auto stage = [&](int cps, unsigned bufoff) {     // issue only; no wait
        const unsigned long long sb = xnbase + (unsigned)(cps << 8);
        for (int it = 0; it < 19; ++it) {
            int t = tid + (it << 8);
            if (t < 4624) {                          // 289 pixels * 16 chunks
                int ch = t & 15, pix = t >> 4;
                int riy = pix / 17, rix = pix - riy * 17;
                int iy = iy0 + riy, ix = ix0 + rix;
                unsigned lds = bufoff + pix * 272 + ch * 16;
                if ((unsigned)iy < 64u && (unsigned)ix < 64u) {
                    unsigned voff = ((unsigned)((iy << 6) + ix) << 10) + (ch << 4);
                    asm volatile("global_load_async_to_lds_b128 %0, %1, %2"
                                 :: "v"(lds), "v"(voff), "s"(sb) : "memory");
                } else {
                    *(int4*)(smem + lds) = make_int4(0, 0, 0, 0);
                }
            }
        }
    };

    stage(0, XS0_OFF);                               // first stage, pre-loop

    for (int occ = 0; occ < 4; ++occ) {
        // zero y tile (overlaps in-flight cp0 stage)
        for (int t = tid; t < YLDS_DW / 4; t += 256)
            ((float4*)yL)[t] = make_float4(0.f, 0.f, 0.f, 0.f);

        v8f accA[16], accB[4];
        #pragma unroll
        for (int a = 0; a < 16; ++a) accA[a] = (v8f){0,0,0,0,0,0,0,0};
        #pragma unroll
        for (int a = 0; a < 4; ++a)  accB[a] = (v8f){0,0,0,0,0,0,0,0};

        const int ocblk = zh * 8 + occ * 2 + oct;

        for (int cp = 0; cp < 4; ++cp) {
            const unsigned cur = cp & 1;
            asm volatile("s_wait_asynccnt 0" ::: "memory");
            __syncthreads();                         // buf[cur] staged for all waves
            if (cp < 3) stage(cp + 1, cur ? XS0_OFF : XS1_OFF);
            const char* xsC = (const char*)smem + (cur ? XS1_OFF : XS0_OFF);

            for (int ks = 0; ks < 4; ++ks) {
                const int cblk = cp * 4 + ks;
                const char* xK = xsC + ks * 64 + laneAoff;

                // ---- segA: 2 taps x 16 rows, 2-deep A prefetch ----
                #pragma unroll
                for (int t = 0; t < 2; ++t) {
                    const int ky = t ? ky1 : ky0;
                    const int kx = t ? kx1 : kx0;
                    const int tap = ky * 3 + kx;
                    const v8h* bp = (const v8h*)(wp +
                        (size_t)((((tap * 16 + cblk) * 16 + ocblk) * 32 + ln) * 16));
                    const v16h B = cat16(bp[0], bp[1]);
                    const int offy = (py - ky + 2) >> 1;
                    const int offx = (px - kx + 2) >> 1;
                    const char* xb = xK + offx * (XPITCH * 2) + offy * ROWSTEP;
                    v8h c0l = *(const v8h*)xb, c0h = *(const v8h*)(xb + 32);
                    v8h c1l = *(const v8h*)(xb + ROWSTEP), c1h = *(const v8h*)(xb + ROWSTEP + 32);
                    #pragma unroll
                    for (int a = 0; a < 16; ++a) {
                        const v16h A = cat16(c0l, c0h);
                        c0l = c1l; c0h = c1h;
                        if (a < 14) {
                            const char* apn = xb + (a + 2) * ROWSTEP;
                            c1l = *(const v8h*)apn; c1h = *(const v8h*)(apn + 32);
                        }
                        accA[a] = __builtin_amdgcn_wmma_f32_16x16x32_f16(
                            false, A, false, B, (short)0, accA[a], false, false);
                    }
                }
                // ---- segB: oo tap (1,1), 4 rows ----
                {
                    const v8h* bp = (const v8h*)(wp +
                        (size_t)((((4 * 16 + cblk) * 16 + ocblk) * 32 + ln) * 16));
                    const v16h B = cat16(bp[0], bp[1]);
                    const char* xb = xK + (XPITCH * 2) + (1 + rB0) * ROWSTEP;
                    v8h c0l = *(const v8h*)xb, c0h = *(const v8h*)(xb + 32);
                    v8h c1l = *(const v8h*)(xb + ROWSTEP), c1h = *(const v8h*)(xb + ROWSTEP + 32);
                    #pragma unroll
                    for (int a = 0; a < 4; ++a) {
                        const v16h A = cat16(c0l, c0h);
                        c0l = c1l; c0h = c1h;
                        if (a < 2) {
                            const char* apn = xb + (a + 2) * ROWSTEP;
                            c1l = *(const v8h*)apn; c1h = *(const v8h*)(apn + 32);
                        }
                        accB[a] = __builtin_amdgcn_wmma_f32_16x16x32_f16(
                            false, A, false, B, (short)0, accB[a], false, false);
                    }
                }
            }
        }

        // ---- flush accumulators into y tile (ds_add_f32 merges tap-split waves)
        #pragma unroll
        for (int a = 0; a < 16; ++a) {
            const int oy = 2 * a + py;
            #pragma unroll
            for (int v = 0; v < 8; ++v) {
                const int m = v + hiL * 8;
                atomicAdd(&yL[oy * YROW + (2 * m + px) * YP + ocl], accA[a][v]);
            }
        }
        #pragma unroll
        for (int a = 0; a < 4; ++a) {
            const int oy = 2 * (rB0 + a) + 1;
            #pragma unroll
            for (int v = 0; v < 8; ++v) {
                const int m = v + hiL * 8;
                atomicAdd(&yL[oy * YROW + (2 * m + 1) * YP + ocl], accB[a][v]);
            }
        }
        __syncthreads();                 // all partial sums in yL; both xs free

        // Next oc-chunk's cp0 stage overlaps the whole FIR phase (targets xs0;
        // h-pass lives in xs1 as f16, y is read-only from here on).
        if (occ < 3) stage(0, XS0_OFF);

        // ---- FIR pass 1 (horizontal, rolling window): lanes->oc, conflict-free
        {
            const int ocF = tid & 31;
            const int rb  = tid >> 5;
            for (int k = 0; k < 4; ++k) {
                int hy = rb + 8 * k;                 // 0..31 (31 valid)
                if (hy < 31) {
                    const float* yrow = yL + (hy + 1) * YROW + ocF;
                    float w1 = yrow[1 * YP], w2 = yrow[2 * YP], w3 = yrow[3 * YP];
                    _Float16* hrow = hH + (ocF * 31 + hy) * 28;
                    #pragma unroll 4
                    for (int j = 0; j < 28; ++j) {
                        float w4 = yrow[(j + 4) * YP];
                        hrow[j] = (_Float16)((w1 + w4) + 3.f * (w2 + w3));
                        w1 = w2; w2 = w3; w3 = w4;
                    }
                }
            }
        }
        __syncthreads();

        // ---- FIR pass 2 (vertical, rolling) + bias/leaky/clamp, coalesced store
        {
            const int jF = tid & 31;
            for (int k = 0; k < 4; ++k) {
                int oc = (tid >> 5) + 8 * k;         // wave-uniform
                if (jF < 28) {
                    const int ocAbs = zh * 128 + occ * 32 + oc;
                    const float b = bias[ocAbs];
                    const _Float16* hcol = hH + (oc * 31) * 28 + jF;
                    float w1 = (float)hcol[0], w2 = (float)hcol[28], w3 = (float)hcol[56];
                    const int oj = c0 + jF;
                    #pragma unroll 4
                    for (int i = 0; i < 28; ++i) {
                        float w4 = (float)hcol[(i + 3) * 28];
                        float s = ((w1 + w4) + 3.f * (w2 + w3)) * 0.0625f + b;
                        w1 = w2; w2 = w3; w3 = w4;
                        s = (s >= 0.f ? s : 0.2f * s) * ACTGAIN;
                        s = fminf(fmaxf(s, -256.f), 256.f);
                        const int oi = r0 + i;
                        if (oi < OH && oj < OW)
                            out[(((size_t)n * OC + ocAbs) * OH + oi) * OW + oj] = s;
                    }
                }
            }
        }
        __syncthreads();                 // h consumed; xs1 reusable next chunk
    }
}

// ---------------------------------------------------------------------------
extern "C" void kernel_launch(void* const* d_in, const int* in_sizes, int n_in,
                              void* d_out, int out_size, void* d_ws, size_t ws_size,
                              hipStream_t stream) {
    const float* x    = (const float*)d_in[0];
    const float* wt   = (const float*)d_in[1];
    const float* bias = (const float*)d_in[2];
    float* outp       = (float*)d_out;

    _Float16* wp  = (_Float16*)d_ws;                         // 2,359,296 B
    _Float16* x16 = (_Float16*)((char*)d_ws + 2359296);      // 67,108,864 B

    hipFuncSetAttribute(reinterpret_cast<const void*>(prep_x_kernel),
                        hipFuncAttributeMaxDynamicSharedMemorySize, SMEM_PREPX);
    hipFuncSetAttribute(reinterpret_cast<const void*>(conv_up_fir_kernel),
                        hipFuncAttributeMaxDynamicSharedMemorySize, SMEM_MAIN);

    prep_w_kernel<<<2304, 256, 0, stream>>>(wt, wp);
    prep_x_kernel<<<dim3(64, 16), 256, SMEM_PREPX, stream>>>(x, x16);
    conv_up_fir_kernel<<<dim3(25, 16, 2), 256, SMEM_MAIN, stream>>>(x16, wp, bias, outp);
}
